// GraphWaveNetPredictor_84954453115331
// MI455X (gfx1250) — compile-verified, matrix operand order checked by reading, not run on hardware
//
#include <hip/hip_runtime.h>
#include <hip/hip_bf16.h>
#include <math.h>

// GraphWaveNet forward on MI455X (gfx1250, wave32, WMMA).
// Heavy GEMMs run as v_wmma_f32_16x16x32_bf16 with f32 accumulation:
//  * gated conv: (T*B*N x 128) @ (128 x 128)  [concat h(t-d)|h(t), concat Wf|Wg]
//  * graph conv: per (t,b): A(512x512 padded) @ hg(512x64 padded)
// bf16 storage halves HBM traffic (23.3 TB/s roofline) vs f32.
//
// Activations are stored (t, b, n, c) with a zeroed 256000-row pad BEFORE row 0,
// so the dilated tap h[t-d] is an unconditional load at row - d*B*N: causal
// zero-padding falls out of the layout with no predication. Every WMMA fragment
// is two aligned 128-bit loads (per the ISA 16-bit 16x32 VGPR layout, a lane's
// 16 bf16 values are two contiguous 16-byte runs at K = khalf and khalf+16).
// Gate nonlinearities are branchless (clamped exp) so the epilogue stays
// straight-line VALU with no exec-mask divergence.

typedef __attribute__((ext_vector_type(16))) __bf16 v16bf;
typedef __attribute__((ext_vector_type(8)))  float  v8f;
typedef __attribute__((ext_vector_type(4)))  unsigned int v4u;

#define Bdim 16
#define Tdim 64
#define Ndim 500
#define NP   512
#define Hdim 64
#define Sdim 64
#define Odim 3
#define Ldim 6
#define ROWS (Bdim*Tdim*Ndim)        // 512000, flattened (t,b,n)
#define RPT  (Bdim*Ndim)             // rows per time step = 8000
#define PADROWS (32*RPT)             // max dilation (32) worth of zero rows

union Frag { v16bf v; v4u q[2]; };

__device__ __forceinline__ float sigm(float x) {
    return 1.0f / (1.0f + __expf(-x));
}

// Branchless tanh: 1 - 2/(e^{2x}+1), x clamped so the exp stays finite.
__device__ __forceinline__ float tanh_fast(float x) {
    float xc = fminf(fmaxf(x, -15.0f), 15.0f);
    return 1.0f - 2.0f / (__expf(2.0f * xc) + 1.0f);
}

// ---------------------------------------------------------------------------
__global__ void zero4_kernel(v4u* __restrict__ p, long n16) {
    long i = (long)blockIdx.x * blockDim.x + threadIdx.x;
    if (i < n16) { v4u z = {0u, 0u, 0u, 0u}; p[i] = z; }
}

// ---------------------------------------------------------------------------
// Adjacency: A = rownorm(adj + I) + softmax(relu(nv1 @ nv2^T), axis=1),
// written as bf16 padded to 512x512 (zeros outside 500x500).
__global__ void build_A_kernel(const float* __restrict__ adj,
                               const float* __restrict__ nv1,
                               const float* __restrict__ nv2,
                               __bf16* __restrict__ A_bf) {
    __shared__ float red[NP];
    __shared__ float rowsum_s, maxe_s, sume_s;
    int m = blockIdx.x;   // 0..511
    int n = threadIdx.x;  // 0..511
    if (m >= Ndim) { A_bf[(long)m * NP + n] = (__bf16)0.0f; return; }

    float e = -1e30f, adjv = 0.0f;
    if (n < Ndim) {
        float dot = 0.0f;
        #pragma unroll
        for (int k = 0; k < 10; ++k) dot += nv1[m * 10 + k] * nv2[n * 10 + k];
        e = fmaxf(dot, 0.0f);
        adjv = adj[m * Ndim + n] + ((m == n) ? 1.0f : 0.0f);
    }
    red[n] = (n < Ndim) ? adjv : 0.0f; __syncthreads();
    for (int s = NP / 2; s > 0; s >>= 1) { if (n < s) red[n] += red[n + s]; __syncthreads(); }
    if (n == 0) { float rs = red[0]; rowsum_s = (rs == 0.0f) ? 1.0f : rs; } __syncthreads();

    red[n] = (n < Ndim) ? e : -1e30f; __syncthreads();
    for (int s = NP / 2; s > 0; s >>= 1) { if (n < s) red[n] = fmaxf(red[n], red[n + s]); __syncthreads(); }
    if (n == 0) maxe_s = red[0]; __syncthreads();

    float ex = (n < Ndim) ? __expf(e - maxe_s) : 0.0f;
    red[n] = ex; __syncthreads();
    for (int s = NP / 2; s > 0; s >>= 1) { if (n < s) red[n] += red[n + s]; __syncthreads(); }
    if (n == 0) sume_s = red[0]; __syncthreads();

    float val = (n < Ndim) ? (adjv / rowsum_s + ex / sume_s) : 0.0f;
    A_bf[(long)m * NP + n] = (__bf16)val;
}

// ---------------------------------------------------------------------------
// WcatT[l][o][k] bf16, o in 0..127 (f then g outputs), k in 0..127 (x0 tap | h tap).
// k-contiguous so B fragments are two b128 loads.
__global__ void prep_weights_kernel(const float* __restrict__ Wf,
                                    const float* __restrict__ Wg,
                                    __bf16* __restrict__ WcatT) {
    int idx = blockIdx.x * blockDim.x + threadIdx.x;   // l*16384 + o*128 + k
    if (idx >= Ldim * 128 * 128) return;
    int l = idx >> 14, rem = idx & 16383, o = rem >> 7, k = rem & 127;
    int tap = (k < 64) ? 0 : 1;
    int ki = k & 63;
    float w;
    if (o < 64) w = Wf[((l * 64 + o) * 64 + ki) * 2 + tap];
    else        w = Wg[((l * 64 + (o - 64)) * 64 + ki) * 2 + tap];
    WcatT[idx] = (__bf16)w;
}

// ---------------------------------------------------------------------------
// h = relu(x * W_in + b_in) : f32 master + bf16 copy, in (t,b,n,c) layout.
__global__ void input_proj_kernel(const float* __restrict__ x,
                                  const float* __restrict__ W_in,
                                  const float* __restrict__ b_in,
                                  float* __restrict__ h_f32,
                                  __bf16* __restrict__ h_bf) {
    long idx = (long)blockIdx.x * blockDim.x + threadIdx.x;
    if (idx >= (long)ROWS * Hdim) return;
    long row = idx >> 6; int c = (int)(idx & 63);
    int tb = (int)(row / Ndim), n = (int)(row % Ndim);
    int t = tb >> 4, b = tb & 15;                 // tb = t*16 + b
    float xv = x[((long)(b * Tdim + t)) * Ndim + n];   // x is (b,t,n)
    float v = fmaxf(xv * W_in[c] + b_in[c], 0.0f);
    h_f32[idx] = v;
    h_bf[idx]  = (__bf16)v;
}

// ---------------------------------------------------------------------------
// Gated dilated conv: M = T*B*N rows, K = 128 ([h(t-d) | h(t)]), out = 128
// ([f | g]). Block = 128 thr = 4 waves; wave w owns f-cols 16w.. and g-cols
// 64+16w.., so tanh*sigmoid fuses in-register. hg stored transposed per (t,b):
// hgT[tb][c][n] (n contiguous = K dim of the graph conv).
__global__ void gated_conv_kernel(const __bf16* __restrict__ h_bf,   // valid rows at >= 0; zero pad below
                                  const __bf16* __restrict__ WcatT,  // [128][128]
                                  const float* __restrict__ bfv,
                                  const float* __restrict__ bgv,
                                  __bf16* __restrict__ hgT,
                                  int d) {
    int tile = blockIdx.x;                 // 16 rows per tile
    int wave = threadIdx.x >> 5;           // 0..3
    int lane = threadIdx.x & 31;
    int r    = tile * 16 + (lane & 15);    // flattened (t,b,n) row
    int khalf = (lane >> 4) << 3;          // 0 or 8 (element offset)
    int cf = wave * 16 + (lane & 15);      // f column (0..63)
    int cg = 64 + cf;                      // g column (64..127)

    const __bf16* hrow = h_bf + (long)r * Hdim;
    const __bf16* xrow = h_bf + (long)(r - d * RPT) * Hdim;  // zero pad handles t<d
    const __bf16* wF = WcatT + (long)cf * 128;
    const __bf16* wG = WcatT + (long)cg * 128;

    v8f accF = {}, accG = {};
    #pragma unroll
    for (int kb = 0; kb < 128; kb += 32) {
        const __bf16* abase = (kb < 64) ? (xrow + kb) : (hrow + (kb - 64));
        Frag a, bF, bG;
        a.q[0]  = *(const v4u*)(abase + khalf);
        a.q[1]  = *(const v4u*)(abase + khalf + 16);
        bF.q[0] = *(const v4u*)(wF + kb + khalf);
        bF.q[1] = *(const v4u*)(wF + kb + khalf + 16);
        bG.q[0] = *(const v4u*)(wG + kb + khalf);
        bG.q[1] = *(const v4u*)(wG + kb + khalf + 16);
        accF = __builtin_amdgcn_wmma_f32_16x16x32_bf16(false, a.v, false, bF.v, (short)0, accF, false, false);
        accG = __builtin_amdgcn_wmma_f32_16x16x32_bf16(false, a.v, false, bG.v, (short)0, accG, false, false);
    }
    // Epilogue: C/D layout -> lane 0-15: N=lane, M=v ; lane 16-31: M=v+8.
    int mhalf = (lane >> 4) ? 8 : 0;
    float bfc = bfv[cf], bgc = bgv[cf];
    #pragma unroll
    for (int v = 0; v < 8; ++v) {
        int rr = tile * 16 + v + mhalf;
        int tbv = rr / Ndim;
        int nv  = rr % Ndim;
        float f = accF[v] + bfc;
        float g = accG[v] + bgc;
        float hg = tanh_fast(f) * sigm(g);
        hgT[((long)tbv * Hdim + cf) * NP + nv] = (__bf16)hg;
    }
}

// ---------------------------------------------------------------------------
// Skip: skip[b][n][s] += sum_c hg(t=T-1,b,n,c) * Ws[s][c] + bs[s]
// One block per (b,n); hg vector staged in LDS.
__global__ void skip_accum_kernel(const __bf16* __restrict__ hgT,
                                  const float* __restrict__ Ws,
                                  const float* __restrict__ bs,
                                  float* __restrict__ skip) {
    __shared__ float hv[Hdim];
    int bn = blockIdx.x;                  // 0..7999 = b*500 + n
    int s  = threadIdx.x;                 // 0..63
    int b = bn / Ndim, n = bn % Ndim;
    int tb = (Tdim - 1) * Bdim + b;
    hv[s] = (float)hgT[((long)tb * Hdim + s) * NP + n];
    __syncthreads();
    float acc = bs[s];
    #pragma unroll 8
    for (int c = 0; c < Hdim; ++c) acc += hv[c] * Ws[s * Hdim + c];
    skip[(long)bn * Sdim + s] += acc;
}

// ---------------------------------------------------------------------------
// Graph conv + residual: per (t,b): Hs = A(512x512) @ hgT^T(512x64);
// h_new = last ? Hs : h_old + relu(Hs). Block = 256 thr = 8 waves covering
// 2 M-tiles x 4 C-tiles; K loop 512/32 = 16 WMMAs per wave. hgT pad columns
// (n=500..511) are pre-zeroed so no guard is needed.
__global__ void graph_conv_kernel(const __bf16* __restrict__ A_bf,   // [512][512]
                                  const __bf16* __restrict__ hgT,    // [1024][64][512]
                                  float* __restrict__ h_f32,
                                  __bf16* __restrict__ h_bf,
                                  int last) {
    int tb   = blockIdx.y;               // 0..1023
    int mtb  = blockIdx.x;               // 0..15
    int wave = threadIdx.x >> 5;         // 0..7
    int lane = threadIdx.x & 31;
    int mt   = mtb * 2 + (wave >> 2);    // M tile 0..31
    int ct   = wave & 3;                 // C tile 0..3
    int mrow = mt * 16 + (lane & 15);
    int col  = ct * 16 + (lane & 15);
    int khalf = (lane >> 4) << 3;

    const __bf16* Arow = A_bf + (long)mrow * NP;
    const __bf16* Bcol = hgT + ((long)tb * Hdim + col) * NP;
    v8f acc = {};
    #pragma unroll 4
    for (int kb = 0; kb < NP; kb += 32) {
        Frag a, b;
        a.q[0] = *(const v4u*)(Arow + kb + khalf);
        a.q[1] = *(const v4u*)(Arow + kb + khalf + 16);
        b.q[0] = *(const v4u*)(Bcol + kb + khalf);
        b.q[1] = *(const v4u*)(Bcol + kb + khalf + 16);
        acc = __builtin_amdgcn_wmma_f32_16x16x32_bf16(false, a.v, false, b.v, (short)0, acc, false, false);
    }
    int mhalf = (lane >> 4) ? 8 : 0;
    #pragma unroll
    for (int v = 0; v < 8; ++v) {
        int m = mt * 16 + v + mhalf;
        if (m < Ndim) {
            long idx = ((long)tb * Ndim + m) * Hdim + col;
            float hs = acc[v];
            float hn = last ? hs : (h_f32[idx] + fmaxf(hs, 0.0f));
            h_f32[idx] = hn;
            h_bf[idx]  = (__bf16)hn;
        }
    }
}

// ---------------------------------------------------------------------------
// Head: out[b][o][n] = relu(relu(skip) @ W1^T + b1) @ W2^T + b2
__global__ void head_kernel(const float* __restrict__ skip,
                            const float* __restrict__ W1, const float* __restrict__ b1,
                            const float* __restrict__ W2, const float* __restrict__ b2,
                            float* __restrict__ out) {
    __shared__ float sv[Hdim];
    __shared__ float h1[Hdim];
    int bn = blockIdx.x;                 // 0..7999
    int j  = threadIdx.x;                // 0..63
    int b = bn / Ndim, n = bn % Ndim;
    sv[j] = fmaxf(skip[(long)bn * Sdim + j], 0.0f);
    __syncthreads();
    float acc = b1[j];
    #pragma unroll 8
    for (int c = 0; c < Sdim; ++c) acc += W1[j * Sdim + c] * sv[c];
    h1[j] = fmaxf(acc, 0.0f);
    __syncthreads();
    if (j < Odim) {
        float o = b2[j];
        #pragma unroll 8
        for (int c = 0; c < Hdim; ++c) o += W2[j * Hdim + c] * h1[c];
        out[((long)b * Odim + j) * Ndim + n] = o;
    }
}

// ---------------------------------------------------------------------------
extern "C" void kernel_launch(void* const* d_in, const int* in_sizes, int n_in,
                              void* d_out, int out_size, void* d_ws, size_t ws_size,
                              hipStream_t stream) {
    const float* x    = (const float*)d_in[0];
    const float* adj  = (const float*)d_in[1];
    const float* nv1  = (const float*)d_in[2];
    const float* nv2  = (const float*)d_in[3];
    const float* W_in = (const float*)d_in[4];
    const float* b_in = (const float*)d_in[5];
    const float* Wf   = (const float*)d_in[6];
    const float* bf_  = (const float*)d_in[7];
    const float* Wg   = (const float*)d_in[8];
    const float* bg_  = (const float*)d_in[9];
    const float* Ws   = (const float*)d_in[10];
    const float* bs   = (const float*)d_in[11];
    const float* W1   = (const float*)d_in[12];
    const float* b1   = (const float*)d_in[13];
    const float* W2   = (const float*)d_in[14];
    const float* b2   = (const float*)d_in[15];
    float* out = (float*)d_out;

    char* ws = (char*)d_ws;
    size_t off = 0;
    auto alloc = [&](size_t bytes) -> void* {
        void* p = ws + off;
        off = (off + bytes + 255) & ~(size_t)255;
        return p;
    };
    float*  h_f32    = (float*) alloc((size_t)ROWS * Hdim * 4);               // 131 MB
    __bf16* h_bf_pad = (__bf16*)alloc((size_t)(PADROWS + ROWS) * Hdim * 2);   //  98 MB
    __bf16* h_bf     = h_bf_pad + (size_t)PADROWS * Hdim;
    __bf16* hgT      = (__bf16*)alloc((size_t)Bdim * Tdim * Hdim * NP * 2);   //  64 MB
    __bf16* A_bf     = (__bf16*)alloc((size_t)NP * NP * 2);                   // 512 KB
    __bf16* WcatT    = (__bf16*)alloc((size_t)Ldim * 128 * 128 * 2);
    float*  skip     = (float*) alloc((size_t)Bdim * Ndim * Sdim * 4);        //   2 MB
    (void)ws_size; (void)in_sizes; (void)n_in; (void)out_size;

    // Zero: h pad rows (causal padding), full hgT (K-pad cols), skip accumulator.
    long padQ  = (long)PADROWS * Hdim * 2 / 16;
    long hgtQ  = (long)Bdim * Tdim * Hdim * NP * 2 / 16;
    long skipQ = (long)Bdim * Ndim * Sdim * 4 / 16;
    zero4_kernel<<<(unsigned)((padQ  + 255) / 256), 256, 0, stream>>>((v4u*)h_bf_pad, padQ);
    zero4_kernel<<<(unsigned)((hgtQ  + 255) / 256), 256, 0, stream>>>((v4u*)hgT, hgtQ);
    zero4_kernel<<<(unsigned)((skipQ + 255) / 256), 256, 0, stream>>>((v4u*)skip, skipQ);

    build_A_kernel<<<NP, NP, 0, stream>>>(adj, nv1, nv2, A_bf);
    prep_weights_kernel<<<(Ldim * 128 * 128 + 255) / 256, 256, 0, stream>>>(Wf, Wg, WcatT);
    long tot = (long)ROWS * Hdim;
    input_proj_kernel<<<(unsigned)((tot + 255) / 256), 256, 0, stream>>>(x, W_in, b_in, h_f32, h_bf);

    for (int l = 0; l < Ldim; ++l) {
        int d = 1 << l;
        gated_conv_kernel<<<ROWS / 16, 128, 0, stream>>>(
            h_bf, WcatT + (size_t)l * 128 * 128, bf_ + l * Hdim, bg_ + l * Hdim, hgT, d);
        skip_accum_kernel<<<Bdim * Ndim, Sdim, 0, stream>>>(
            hgT, Ws + (size_t)l * Sdim * Hdim, bs + l * Sdim, skip);
        dim3 g(16, Bdim * Tdim);
        graph_conv_kernel<<<g, 256, 0, stream>>>(A_bf, hgT, h_f32, h_bf, (l == Ldim - 1) ? 1 : 0);
    }
    head_kernel<<<Bdim * Ndim, Sdim, 0, stream>>>(skip, W1, b1, W2, b2, out);
}
